// ISDAHead_18021682774196
// MI455X (gfx1250) — matrix operand order repaired and checked by reading, not compiled, not run on hardware
//
#include <hip/hip_runtime.h>

// Problem constants (from reference)
#define N_  16384
#define A_  768
#define C_  1000
#define CP  1024          // C padded to multiple of 16 for WMMA tiles
#define K_  1536          // 2*A : concatenated-K fused GEMM
#define KB  48            // K_/32 k-blocks for 16x16x32 bf16 WMMA
#define MT  64            // CP/16 row/col tiles

typedef __attribute__((ext_vector_type(16))) __bf16 v16bf;
typedef __attribute__((ext_vector_type(8)))  float  v8f;

// ---------------------------------------------------------------- K0: zero
__global__ void isda_zero(float* __restrict__ p, int n) {
    int i = blockIdx.x * blockDim.x + threadIdx.x;
    int stride = gridDim.x * blockDim.x;
    for (; i < n; i += stride) p[i] = 0.0f;
}

// ------------------------------------------------- K1: per-class sums (atomics)
// counts[c] = #samples; sumf[c,a] = sum f; sumf2[c,a] = sum f^2
__global__ void isda_accum(const float* __restrict__ features,
                           const int*   __restrict__ labels,
                           float* __restrict__ counts,
                           float* __restrict__ sumf,
                           float* __restrict__ sumf2) {
    const int per_row = A_ / 4;                       // 192 float4 per row
    int idx = blockIdx.x * blockDim.x + threadIdx.x;
    if (idx >= N_ * per_row) return;
    int n = idx / per_row;
    int j = idx - n * per_row;
    int l = labels[n];
    float4 f = ((const float4*)features)[idx];
    float* s  = sumf  + (size_t)l * A_ + 4 * j;
    float* s2 = sumf2 + (size_t)l * A_ + 4 * j;
    atomicAdd(s + 0, f.x);  atomicAdd(s + 1, f.y);
    atomicAdd(s + 2, f.z);  atomicAdd(s + 3, f.w);
    atomicAdd(s2 + 0, f.x * f.x);  atomicAdd(s2 + 1, f.y * f.y);
    atomicAdd(s2 + 2, f.z * f.z);  atomicAdd(s2 + 3, f.w * f.w);
    if (j == 0) atomicAdd(&counts[l], 1.0f);
}

// ---------------- K2: finalize new_cov, build WMMA-swizzled bf16 A/B, term3
// A-pack (CPxK bf16): row c = [ncov[c,:] | W[c,:]*ncov[c,:]]
// B-pack (KxCP bf16): col c = [W[c,:]^2 ; -2*W[c,:]]
// Both stored directly in per-wave WMMA fragment layout:
//   A frag (16-bit A 16x32): lane=(m&15)+16h, elems 0-7 -> K=kb*32+8h+0..7,
//                            elems 8-15 -> K=kb*32+16+8h+0..7
//   B frag (16-bit B 32x16): same K mapping as A frag, lane=(n&15)+16h
__global__ void isda_build(const float* __restrict__ fcw,
                           const float* __restrict__ count_in,
                           const float* __restrict__ mean,
                           const float* __restrict__ cov,
                           const float* __restrict__ counts,
                           const float* __restrict__ sumf,
                           const float* __restrict__ sumf2,
                           __bf16* __restrict__ Apk,
                           __bf16* __restrict__ Bpk,
                           float*  __restrict__ term3) {
    int idx = blockIdx.x * blockDim.x + threadIdx.x;
    if (idx >= CP * A_) return;
    int c = idx / A_;
    int a = idx - c * A_;

    float ncov = 0.0f, Wca = 0.0f;
    if (c < C_) {
        float cnt    = counts[c];
        float amount = fmaxf(cnt, 1.0f);
        float sf  = sumf [(size_t)c * A_ + a];
        float sf2 = sumf2[(size_t)c * A_ + a];
        float ave = sf / amount;
        float var = (sf2 - 2.0f * ave * sf + cnt * ave * ave) / amount;
        float denom = cnt + count_in[c];
        float w = (denom > 0.0f) ? (cnt / fmaxf(denom, 1.0f)) : 0.0f;
        float d = mean[(size_t)c * A_ + a] - ave;
        ncov = cov[(size_t)c * A_ + a] * (1.0f - w) + var * w
             + w * (1.0f - w) * d * d;
        Wca = fcw[(size_t)c * A_ + a];
        atomicAdd(&term3[c], Wca * Wca * ncov);        // fp32-exact term3
    }

    int mt = c >> 4, ml = c & 15;
    // ---- A-pack: k=a -> ncov ; k=A_+a -> Wca*ncov
    {
        int k = a, kb = k >> 5, r = k & 31;
        int h = (r >> 3) & 1;
        int e = (r & 7) | ((r >> 4) << 3);
        Apk[(((size_t)(mt * KB + kb)) * 32 + (ml + 16 * h)) * 16 + e] = (__bf16)ncov;
    }
    {
        int k = A_ + a, kb = k >> 5, r = k & 31;
        int h = (r >> 3) & 1;
        int e = (r & 7) | ((r >> 4) << 3);
        Apk[(((size_t)(mt * KB + kb)) * 32 + (ml + 16 * h)) * 16 + e] = (__bf16)(Wca * ncov);
    }
    // ---- B-pack: k=a -> W^2 ; k=A_+a -> -2W   (column c)
    {
        int k = a, kb = k >> 5, r = k & 31;
        int h = (r >> 3) & 1, e = (r & 7) | ((r >> 4) << 3);
        Bpk[(((size_t)(mt * KB + kb)) * 32 + (ml + 16 * h)) * 16 + e] = (__bf16)(Wca * Wca);
    }
    {
        int k = A_ + a, kb = k >> 5, r = k & 31;
        int h = (r >> 3) & 1, e = (r & 7) | ((r >> 4) << 3);
        Bpk[(((size_t)(mt * KB + kb)) * 32 + (ml + 16 * h)) * 16 + e] = (__bf16)(-2.0f * Wca);
    }
}

// ---------------------------- K3: T = 0.5*ratio*(Apk @ Bpk + term3[row])
// block = 256 threads (8 waves); wave computes 64x32 (4x2 WMMA tiles)
// grid = (CP/64, CP/256) = (16, 4)
__global__ void __launch_bounds__(256)
isda_gemm(const __bf16* __restrict__ Apk,
          const __bf16* __restrict__ Bpk,
          const float*  __restrict__ term3,
          const float*  __restrict__ ratio,
          float* __restrict__ T) {
    const int lane = threadIdx.x & 31;
    const int wv   = threadIdx.x >> 5;
    const int mtb  = blockIdx.x * 4;                 // 4 row tiles of 16
    const int ntb  = blockIdx.y * 16 + wv * 2;       // 2 col tiles of 16

    v8f zero = {0.f, 0.f, 0.f, 0.f, 0.f, 0.f, 0.f, 0.f};
    v8f acc[4][2];
    #pragma unroll
    for (int r = 0; r < 4; ++r)
        #pragma unroll
        for (int s = 0; s < 2; ++s) acc[r][s] = zero;

    for (int kb = 0; kb < KB; ++kb) {
        v16bf af[4], bf[2];
        #pragma unroll
        for (int r = 0; r < 4; ++r)
            af[r] = *(const v16bf*)(Apk + (((size_t)((mtb + r) * KB + kb)) * 32 + lane) * 16);
        #pragma unroll
        for (int s = 0; s < 2; ++s)
            bf[s] = *(const v16bf*)(Bpk + (((size_t)((ntb + s) * KB + kb)) * 32 + lane) * 16);
        #pragma unroll
        for (int r = 0; r < 4; ++r)
            #pragma unroll
            for (int s = 0; s < 2; ++s)
                acc[r][s] = __builtin_amdgcn_wmma_f32_16x16x32_bf16(
                    /*neg_a=*/false, af[r], /*neg_b=*/false, bf[s],
                    /*c_mod=*/(short)0, acc[r][s],
                    /*reuse_a=*/false, /*reuse_b=*/false);
    }

    const float scale = 0.5f * ratio[0];
    const int h  = lane >> 4;
    const int nl = lane & 15;
    #pragma unroll
    for (int r = 0; r < 4; ++r) {
        #pragma unroll
        for (int s = 0; s < 2; ++s) {
            const int col = (ntb + s) * 16 + nl;
            #pragma unroll
            for (int v = 0; v < 8; ++v) {
                const int row = (mtb + r) * 16 + v + 8 * h;   // D layout: VGPR v -> M=v / v+8
                T[(size_t)row * CP + col] = scale * (acc[r][s][v] + term3[row]);
            }
        }
    }
}

// --------------------------- K4: out[n,c] = y[n,c] + T[labels[n], c]
// one block per row n; 250 float4 lanes cover C=1000
__global__ void isda_final(const float* __restrict__ y,
                           const int*   __restrict__ labels,
                           const float* __restrict__ T,
                           float* __restrict__ out) {
    const int n = blockIdx.x;
    const int t = threadIdx.x;
    if (t >= C_ / 4) return;
    const int l = labels[n];
    float4 yv = ((const float4*)(y + (size_t)n * C_))[t];
    float4 tv = *(const float4*)(T + (size_t)l * CP + 4 * t);
    float4 o;
    o.x = yv.x + tv.x; o.y = yv.y + tv.y;
    o.z = yv.z + tv.z; o.w = yv.w + tv.w;
    ((float4*)(out + (size_t)n * C_))[t] = o;
}

// ------------------------------------------------------------------ launcher
extern "C" void kernel_launch(void* const* d_in, const int* in_sizes, int n_in,
                              void* d_out, int out_size, void* d_ws, size_t ws_size,
                              hipStream_t stream) {
    const float* y        = (const float*)d_in[0];
    const float* features = (const float*)d_in[1];
    const float* fcw      = (const float*)d_in[2];
    const int*   labels   = (const int*)  d_in[3];
    const float* count_in = (const float*)d_in[4];
    const float* mean     = (const float*)d_in[5];
    const float* cov      = (const float*)d_in[6];
    const float* ratio    = (const float*)d_in[7];
    float* out = (float*)d_out;

    // workspace layout
    float* counts = (float*)d_ws;                 // CP
    float* term3  = counts + CP;                  // CP
    float* sumf   = term3 + CP;                   // C_*A_
    float* sumf2  = sumf + (size_t)C_ * A_;       // C_*A_
    __bf16* Apk   = (__bf16*)(sumf2 + (size_t)C_ * A_);   // MT*KB*32*16 bf16 = 3MB
    __bf16* Bpk   = Apk + (size_t)MT * KB * 32 * 16;      // 3MB
    float* T      = (float*)(Bpk + (size_t)MT * KB * 32 * 16); // CP*CP f32 = 4MB

    // K0: zero counts/term3/sumf/sumf2 (contiguous)
    int nz = 2 * CP + 2 * C_ * A_;
    isda_zero<<<(nz + 255) / 256, 256, 0, stream>>>(counts, nz);

    // K1: per-class accumulation
    int n1 = N_ * (A_ / 4);
    isda_accum<<<(n1 + 255) / 256, 256, 0, stream>>>(features, labels, counts, sumf, sumf2);

    // K2: finalize estimator, emit packed bf16 GEMM operands + term3
    int n2 = CP * A_;
    isda_build<<<(n2 + 255) / 256, 256, 0, stream>>>(fcw, count_in, mean, cov,
                                                     counts, sumf, sumf2,
                                                     Apk, Bpk, term3);

    // K3: 1024x1024x1536 bf16 WMMA GEMM -> T (with +term3, x0.5*ratio folded)
    isda_gemm<<<dim3(CP / 64, CP / 256), 256, 0, stream>>>(Apk, Bpk, term3, ratio, T);

    // K4: streaming epilogue
    isda_final<<<N_, 256, 0, stream>>>(y, labels, T, out);
}
